// AttentionTSSA_22411139350669
// MI455X (gfx1250) — matrix operand order repaired and checked by reading, not compiled, not run on hardware
//
#include <hip/hip_runtime.h>
#include <hip/hip_bf16.h>
#include <math.h>

// ---------------------------------------------------------------------------
// TSSA attention pipeline for MI455X (gfx1250, wave32, WMMA)
//   GEMMs on v_wmma_f32_16x16x32_bf16, everything else bandwidth-shaped.
// Workspace layout (bytes):
//   wbuf  : B*N*C f32        = 268435456
//   Pi    : B*H*N f32        =   4194304
//   norm  : B*C  f32         =     16384
//   attnw : B*C  f32         =     16384
//   S     : B*H  f32         =       256
// ---------------------------------------------------------------------------

typedef __attribute__((ext_vector_type(16))) __bf16 v16bf;
typedef __attribute__((ext_vector_type(8)))  float  v8f;

#define BATCH  8
#define CDIM   512
#define NHEADS 8
#define HDIM   64
#define NTOK   16384   // 128*128

#define TM 128
#define TN 128
#define TK 32
#define APITCH 40      // padded K/M pitch for bf16 LDS tiles

// ---- WMMA fragment loaders (contiguous k order, same for A and B) ----------------
// A: 16x32 bf16, one M-row per lane; lanes 0-15 hold K0..15,
//    lanes 16-31 hold K16..31.  LDS layout As[m][k].
__device__ inline v16bf load_a_frag(const __bf16 (*As)[APITCH], int m, int hi) {
  v16bf a;
#pragma unroll
  for (int e = 0; e < 16; ++e) a[e] = As[m][hi * 16 + e];
  return a;
}
// B: 32x16 bf16, one N-column per lane; lanes 0-15 hold K0..15,
//    lanes 16-31 hold K16..31.  LDS layout Bs[nn][k].
__device__ inline v16bf load_b_frag(const __bf16 (*Bs)[APITCH], int nn, int hi) {
  v16bf b;
#pragma unroll
  for (int e = 0; e < 16; ++e) b[e] = Bs[nn][hi * 16 + e];
  return b;
}

// ---------------------------------------------------------------------------
// GEMM1: wbuf[b][n][c] = sum_k x[b][k][n] * qkv_w[c][k]
// ---------------------------------------------------------------------------
__global__ __launch_bounds__(256)
void gemm1_kernel(const float* __restrict__ x, const float* __restrict__ qkv_w,
                  float* __restrict__ wbuf) {
  __shared__ __bf16 As[TM][APITCH];
  __shared__ __bf16 Bs[TN][APITCH];

  const int n0 = blockIdx.x * TM;
  const int c0 = blockIdx.y * TN;
  const int b  = blockIdx.z;
  const int t    = threadIdx.x;
  const int lane = t & 31, wid = t >> 5;
  const int wm = wid & 3, wn = wid >> 2;    // 4x2 wave grid -> 32x64 per wave
  const int col = lane & 15, hi = lane >> 4;

  v8f acc[2][4];
#pragma unroll
  for (int i = 0; i < 2; ++i)
#pragma unroll
    for (int j = 0; j < 4; ++j)
#pragma unroll
      for (int r = 0; r < 8; ++r) acc[i][j][r] = 0.0f;

  const float* xb = x + (size_t)b * CDIM * NTOK;

  for (int k0 = 0; k0 < CDIM; k0 += TK) {
    // Stage A (transpose: n contiguous in memory -> As[m][k])
    {
      const int k  = t >> 3;          // 0..31
      const int mb = (t & 7) * 16;    // 0..112
      const float* src = xb + (size_t)(k0 + k) * NTOK + n0 + mb;
      if (k0 + TK < CDIM)
        __builtin_prefetch(src + (size_t)TK * NTOK, 0, 1);
#pragma unroll
      for (int i = 0; i < 16; ++i) As[mb + i][k] = (__bf16)src[i];
    }
    // Stage B: Bs[nn][k] = qkv_w[(c0+nn)*C + k0+k]   (k contiguous)
    {
      const int nn = t >> 1;
      const int kh = (t & 1) * 16;
      const float* src = qkv_w + (size_t)(c0 + nn) * CDIM + k0 + kh;
#pragma unroll
      for (int i = 0; i < 16; ++i) Bs[nn][kh + i] = (__bf16)src[i];
    }
    __syncthreads();

    v16bf af[2], bfrag[4];
#pragma unroll
    for (int fm = 0; fm < 2; ++fm)
      af[fm] = load_a_frag(As, wm * 32 + fm * 16 + col, hi);
#pragma unroll
    for (int fn = 0; fn < 4; ++fn)
      bfrag[fn] = load_b_frag(Bs, wn * 64 + fn * 16 + col, hi);

#pragma unroll
    for (int fm = 0; fm < 2; ++fm)
#pragma unroll
      for (int fn = 0; fn < 4; ++fn)
        acc[fm][fn] = __builtin_amdgcn_wmma_f32_16x16x32_bf16(
            false, af[fm], false, bfrag[fn], (short)0, acc[fm][fn], false, false);
    __syncthreads();
  }

  // Epilogue: wbuf[b][n][c], c contiguous across lanes -> coalesced
  float* wout = wbuf + ((size_t)b * NTOK + n0) * CDIM + c0;
#pragma unroll
  for (int fm = 0; fm < 2; ++fm)
#pragma unroll
    for (int fn = 0; fn < 4; ++fn) {
      const int cc = wn * 64 + fn * 16 + col;
#pragma unroll
      for (int r = 0; r < 8; ++r) {
        const int m = wm * 32 + fm * 16 + r + hi * 8;
        wout[(size_t)m * CDIM + cc] = acc[fm][fn][r];
      }
    }
}

// ---------------------------------------------------------------------------
// norm[b][c] = max(sqrt(sum_n wbuf[b][n][c]^2), 1e-12)
// ---------------------------------------------------------------------------
__global__ __launch_bounds__(256)
void norm_kernel(const float* __restrict__ wbuf, float* __restrict__ norm) {
  const int b = blockIdx.x;
  const int c = blockIdx.y * 64 + threadIdx.x;
  const int ty = threadIdx.y;
  const float* p = wbuf + (size_t)b * NTOK * CDIM + c;
  float s = 0.f;
  for (int n = ty; n < NTOK; n += 4) {
    float v = p[(size_t)n * CDIM];
    s += v * v;
  }
  __shared__ float red[4][64];
  red[ty][threadIdx.x] = s;
  __syncthreads();
  if (ty == 0) {
    float tot = red[0][threadIdx.x] + red[1][threadIdx.x] +
                red[2][threadIdx.x] + red[3][threadIdx.x];
    norm[b * CDIM + c] = fmaxf(sqrtf(tot), 1e-12f);
  }
}

__global__ void initS_kernel(float* __restrict__ S) {
  if (threadIdx.x < BATCH * NHEADS) S[threadIdx.x] = 0.f;
}

// ---------------------------------------------------------------------------
// Per-token: logits over heads -> softmax(dim=heads) -> Pi; accumulate S[b][h]
// One wave32 per token; lane l covers channels [16l, 16l+16) (single head).
// ---------------------------------------------------------------------------
__global__ __launch_bounds__(256)
void pi_kernel(const float* __restrict__ wbuf, const float* __restrict__ norm,
               const float* __restrict__ temp, float* __restrict__ Pi,
               float* __restrict__ S) {
  const int t = threadIdx.x, lane = t & 31, wid = t >> 5;
  const size_t T = (size_t)blockIdx.x * 8 + wid;   // global token id
  const int b = (int)(T / NTOK);
  const int n = (int)(T % NTOK);

  const float* wr = wbuf + T * CDIM;
  const float* nr = norm + b * CDIM;
  const int d0 = lane * 16;
  float s = 0.f;
#pragma unroll
  for (int i = 0; i < 16; ++i) {
    float v = wr[d0 + i] / nr[d0 + i];
    s += v * v;
  }
  // reduce the 4 lanes of each head (lanes 4h..4h+3)
  s += __shfl_xor(s, 1, 32);
  s += __shfl_xor(s, 2, 32);
  const int h = lane >> 2;
  const float logit = temp[h] * s;

  float mx = logit;
#pragma unroll
  for (int hh = 0; hh < 8; ++hh) mx = fmaxf(mx, __shfl(logit, hh * 4, 32));
  const float e = __expf(logit - mx);
  float se = 0.f;
#pragma unroll
  for (int hh = 0; hh < 8; ++hh) se += __shfl(e, hh * 4, 32);
  const float pi = e / se;

  __shared__ float sacc[NHEADS];
  if (t < NHEADS) sacc[t] = 0.f;
  __syncthreads();
  if ((lane & 3) == 0) {
    Pi[((size_t)b * NHEADS + h) * NTOK + n] = pi;
    atomicAdd(&sacc[h], pi);            // ds_add_f32
  }
  __syncthreads();
  if (t < NHEADS) atomicAdd(&S[b * NHEADS + t], sacc[t]);
}

// ---------------------------------------------------------------------------
// attnw[b][c] = 1 / (1 + (sum_n Pi[b][h][n]*w^2) / (S[b][h]+1e-8))
// ---------------------------------------------------------------------------
__global__ __launch_bounds__(256)
void dots_kernel(const float* __restrict__ wbuf, const float* __restrict__ Pi,
                 const float* __restrict__ S, float* __restrict__ attnw) {
  const int b = blockIdx.x;
  const int c = blockIdx.y * 64 + threadIdx.x;
  const int ty = threadIdx.y;
  const int h = c >> 6;
  const float* p   = wbuf + (size_t)b * NTOK * CDIM + c;
  const float* pip = Pi + ((size_t)b * NHEADS + h) * NTOK;
  float s = 0.f;
  for (int n = ty; n < NTOK; n += 4) {
    float v = p[(size_t)n * CDIM];
    s += pip[n] * v * v;
  }
  __shared__ float red[4][64];
  red[ty][threadIdx.x] = s;
  __syncthreads();
  if (ty == 0) {
    float tot = red[0][threadIdx.x] + red[1][threadIdx.x] +
                red[2][threadIdx.x] + red[3][threadIdx.x];
    float inv = 1.f / (S[b * NHEADS + h] + 1e-8f);
    attnw[b * CDIM + c] = 1.f / (1.f + tot * inv);
  }
}

// ---------------------------------------------------------------------------
// GEMM2: out[b][dout][n] = sum_c (-(w*Pi)*attn)[b][n][c] * out_w[dout][c] + out_b
// Elementwise scaling fused into the A-tile staging; output transposed
// through per-wave LDS so [b][c][n] stores are contiguous in n.
// ---------------------------------------------------------------------------
__global__ __launch_bounds__(256)
void gemm2_kernel(const float* __restrict__ wbuf, const float* __restrict__ Pi,
                  const float* __restrict__ attnw, const float* __restrict__ out_w,
                  const float* __restrict__ out_b, float* __restrict__ out) {
  __shared__ __bf16 As[TM][APITCH];
  __shared__ __bf16 Bs[TN][APITCH];
  __shared__ float  tb[8][16][17];

  const int n0 = blockIdx.x * TM;
  const int c0 = blockIdx.y * TN;   // dout tile
  const int b  = blockIdx.z;
  const int t    = threadIdx.x;
  const int lane = t & 31, wid = t >> 5;
  const int wm = wid & 3, wn = wid >> 2;
  const int col = lane & 15, hi = lane >> 4;

  v8f acc[2][4];
#pragma unroll
  for (int i = 0; i < 2; ++i)
#pragma unroll
    for (int j = 0; j < 4; ++j)
#pragma unroll
      for (int r = 0; r < 8; ++r) acc[i][j][r] = 0.0f;

  for (int k0 = 0; k0 < CDIM; k0 += TK) {
    // Stage A with fused -(w * Pi) * attn scaling; 16-channel span stays in
    // one head (16-aligned, head width 64), so one Pi load per thread.
    {
      const int m  = t >> 1;
      const int kh = (t & 1) * 16;
      const int h  = (k0 + kh) >> 6;
      const float piv = Pi[((size_t)b * NHEADS + h) * NTOK + n0 + m];
      const float* src = wbuf + ((size_t)b * NTOK + n0 + m) * CDIM + k0 + kh;
      const float* aw  = attnw + b * CDIM + k0 + kh;
#pragma unroll
      for (int i = 0; i < 16; ++i)
        As[m][kh + i] = (__bf16)(-src[i] * piv * aw[i]);
    }
    // Stage B: Bs[nn][k] = out_w[(c0+nn)*C + k0+k]
    {
      const int nn = t >> 1;
      const int kh = (t & 1) * 16;
      const float* src = out_w + (size_t)(c0 + nn) * CDIM + k0 + kh;
#pragma unroll
      for (int i = 0; i < 16; ++i) Bs[nn][kh + i] = (__bf16)src[i];
    }
    __syncthreads();

    v16bf af[2], bfrag[4];
#pragma unroll
    for (int fm = 0; fm < 2; ++fm)
      af[fm] = load_a_frag(As, wm * 32 + fm * 16 + col, hi);
#pragma unroll
    for (int fn = 0; fn < 4; ++fn)
      bfrag[fn] = load_b_frag(Bs, wn * 64 + fn * 16 + col, hi);

#pragma unroll
    for (int fm = 0; fm < 2; ++fm)
#pragma unroll
      for (int fn = 0; fn < 4; ++fn)
        acc[fm][fn] = __builtin_amdgcn_wmma_f32_16x16x32_bf16(
            false, af[fm], false, bfrag[fn], (short)0, acc[fm][fn], false, false);
    __syncthreads();
  }

  // Epilogue: transpose each 16x16 tile via per-wave LDS, add bias, store
  // out[b][dout][n] with 8 contiguous n-floats per lane.
#pragma unroll
  for (int fm = 0; fm < 2; ++fm)
    for (int fn = 0; fn < 4; ++fn) {
#pragma unroll
      for (int r = 0; r < 8; ++r)
        tb[wid][col][r + hi * 8] = acc[fm][fn][r];   // tb[dout_local][m_local]
      __syncthreads();
      const int dout  = c0 + wn * 64 + fn * 16 + (lane & 15);
      const int nbase = n0 + wm * 32 + fm * 16 + hi * 8;
      const float bias = out_b[dout];
      float* dst = out + ((size_t)b * CDIM + dout) * NTOK + nbase;
#pragma unroll
      for (int i = 0; i < 8; ++i)
        dst[i] = tb[wid][lane & 15][hi * 8 + i] + bias;
      __syncthreads();
    }
}

// ---------------------------------------------------------------------------
extern "C" void kernel_launch(void* const* d_in, const int* in_sizes, int n_in,
                              void* d_out, int out_size, void* d_ws, size_t ws_size,
                              hipStream_t stream) {
  (void)in_sizes; (void)n_in; (void)out_size; (void)ws_size;
  const float* x     = (const float*)d_in[0];
  const float* qkv_w = (const float*)d_in[1];
  const float* temp  = (const float*)d_in[2];
  const float* out_w = (const float*)d_in[3];
  const float* out_b = (const float*)d_in[4];
  float* out = (float*)d_out;

  char* ws = (char*)d_ws;
  float* wbuf  = (float*)ws;                               // 268435456 B
  float* Pi    = (float*)(ws + 268435456);                 //   4194304 B
  float* norm  = (float*)(ws + 268435456 + 4194304);       //     16384 B
  float* attnw = norm + BATCH * CDIM;                      //     16384 B
  float* S     = attnw + BATCH * CDIM;                     //       256 B

  dim3 gemm_grid(NTOK / TM, CDIM / TN, BATCH);             // 128 x 4 x 8
  gemm1_kernel<<<gemm_grid, 256, 0, stream>>>(x, qkv_w, wbuf);

  dim3 red_grid(BATCH, CDIM / 64);
  dim3 red_blk(64, 4);
  norm_kernel<<<red_grid, red_blk, 0, stream>>>(wbuf, norm);

  initS_kernel<<<1, 64, 0, stream>>>(S);

  pi_kernel<<<(BATCH * NTOK) / 8, 256, 0, stream>>>(wbuf, norm, temp, Pi, S);

  dots_kernel<<<red_grid, red_blk, 0, stream>>>(wbuf, Pi, S, attnw);

  gemm2_kernel<<<gemm_grid, 256, 0, stream>>>(wbuf, Pi, attnw, out_w, out_b, out);
}
